// MLA_40913858461952
// MI455X (gfx1250) — matrix operand order repaired and checked by reading, not compiled, not run on hardware
//
#include <hip/hip_runtime.h>
#include <hip/hip_bf16.h>

typedef __attribute__((ext_vector_type(16))) __bf16 v16bf;
typedef __attribute__((ext_vector_type(8)))  __bf16 v8bf;
typedef __attribute__((ext_vector_type(4)))  __bf16 bf16x4;
typedef __attribute__((ext_vector_type(8)))  float  v8f;

#define MLA_S      2048
#define MLA_H      16
#define MLA_SCALE  0.07216878364870323f   // 192^-0.5

// Per-lane async DMA: 16B global -> LDS, tracked by ASYNCcnt.
#define ASYNC_B128(lds, gaddr)                                              \
    asm volatile("global_load_async_to_lds_b128 %0, %1, off"                \
                 :: "v"(lds), "v"(gaddr) : "memory")

// ---------------------------------------------------------------------------
// WMMA helpers (CDNA5 wave32, v_wmma_f32_16x16x32_bf16)
// ---------------------------------------------------------------------------
__device__ __forceinline__ v8f wmma_bf16(v16bf a, v16bf b, v8f c) {
    return __builtin_amdgcn_wmma_f32_16x16x32_bf16(
        /*neg_a=*/false, a, /*neg_b=*/false, b,
        /*c_mod=*/(short)0, c, /*reuse_a=*/false, /*reuse_b=*/false);
}

// A fragment: 16x32 bf16, rows m0..m0+15, K k0..k0+31 from row-major LDS.
__device__ __forceinline__ v16bf load_a_frag(const __bf16* base, int stride,
                                             int m0, int k0, int lane) {
    int l  = lane & 15;
    int hi = (lane >> 4) & 1;
    const __bf16* p = base + (m0 + l) * stride + k0 + hi * 8;
    v8bf lo = *(const v8bf*)(p);
    v8bf hh = *(const v8bf*)(p + 16);
    v16bf r;
#pragma unroll
    for (int i = 0; i < 8; ++i) { r[i] = lo[i]; r[i + 8] = hh[i]; }
    return r;
}

// B fragment: 32x16 (KxN). Lane l<16: col n0+l, K k0..k0+7 and k0+16..k0+23;
// lane l>=16: col n0+(l-16), K k0+8..k0+15 and k0+24..k0+31.
__device__ __forceinline__ v16bf load_b_frag(const __bf16* base, int stride,
                                             int n0, int k0, int lane) {
    const __bf16* p = base + (n0 + (lane & 15)) * stride + k0 + ((lane >> 4) & 1) * 8;
    v8bf lo = *(const v8bf*)(p);
    v8bf hh = *(const v8bf*)(p + 16);
    v16bf r;
#pragma unroll
    for (int i = 0; i < 8; ++i) { r[i] = lo[i]; r[i + 8] = hh[i]; }
    return r;
}

__device__ __forceinline__ bf16x4 cvt4(float4 v) {
    bf16x4 t;
    t.x = (__bf16)v.x; t.y = (__bf16)v.y; t.z = (__bf16)v.z; t.w = (__bf16)v.w;
    return t;
}

// ---------------------------------------------------------------------------
// Elementwise f32 -> bf16 (n multiple of 1024; one float4 per thread).
// ---------------------------------------------------------------------------
__global__ __launch_bounds__(256) void cvt_f32_bf16(
    const float* __restrict__ s, __bf16* __restrict__ d) {
    size_t i = ((size_t)blockIdx.x * 256 + threadIdx.x) * 4;
    float4 v = *(const float4*)(s + i);
    *(bf16x4*)(d + i) = cvt4(v);
}

// Strided row f32 -> bf16 (compacts kv_c rows: stride 576 -> 512).
// One 128-thread block per row, one float4 per thread (R = 512).
__global__ __launch_bounds__(128) void cvt_rows_bf16(
    const float* __restrict__ s, int stride, __bf16* __restrict__ d, int R) {
    int c = threadIdx.x * 4;
    float4 v = *(const float4*)(s + (size_t)blockIdx.x * stride + c);
    *(bf16x4*)(d + (size_t)blockIdx.x * R + c) = cvt4(v);
}

// ---------------------------------------------------------------------------
// GEMM: C[M,N](f32) = A[M,K](bf16, row stride lda) * W[N,K](bf16)^T
// Block: 256 threads = 8 waves. Tile 128(M) x 64(N), BK=32.
// Tiles are DMA'd into double-buffered LDS with global_load_async_to_lds_b128;
// s_wait_asynccnt 3 overlaps the next tile's DMA with this tile's WMMAs.
// Last iteration is peeled so the steady-state loop is branch-free.
// ---------------------------------------------------------------------------
__device__ __forceinline__ void gemm_tile_compute(
    const __bf16* Ab, const __bf16* Bb, int wm, int wn, int lane,
    v8f acc[2][2]) {
    v16bf a0 = load_a_frag(Ab, 32, wm * 32,      0, lane);
    v16bf a1 = load_a_frag(Ab, 32, wm * 32 + 16, 0, lane);
    v16bf b0 = load_b_frag(Bb, 32, wn * 32,      0, lane);
    v16bf b1 = load_b_frag(Bb, 32, wn * 32 + 16, 0, lane);
    acc[0][0] = wmma_bf16(a0, b0, acc[0][0]);
    acc[0][1] = wmma_bf16(a0, b1, acc[0][1]);
    acc[1][0] = wmma_bf16(a1, b0, acc[1][0]);
    acc[1][1] = wmma_bf16(a1, b1, acc[1][1]);
}

__global__ __launch_bounds__(256) void gemm_bf16_async(
    const __bf16* __restrict__ A, int lda,
    const __bf16* __restrict__ W,
    float* __restrict__ C, int M, int N, int K) {
    __shared__ __bf16 As[2][128][32];
    __shared__ __bf16 Bs[2][64][32];

    const int tid  = threadIdx.x;
    const int lane = tid & 31;
    const int wave = tid >> 5;
    const int wm   = wave >> 1;   // 0..3
    const int wn   = wave & 1;    // 0..1
    const int m0   = blockIdx.y * 128;
    const int n0   = blockIdx.x * 64;

    // Per-thread 16B chunks: A tile = 512 chunks (2/thread), B tile = 256 (1/thread).
    const int rA = tid >> 2;            // 0..63
    const int cA = (tid & 3) * 8;       // 0,8,16,24
    const unsigned ldsA0[2] = { (unsigned)(size_t)&As[0][rA][cA],
                                (unsigned)(size_t)&As[1][rA][cA] };
    const unsigned ldsA1[2] = { (unsigned)(size_t)&As[0][rA + 64][cA],
                                (unsigned)(size_t)&As[1][rA + 64][cA] };
    const unsigned ldsB[2]  = { (unsigned)(size_t)&Bs[0][rA][cA],
                                (unsigned)(size_t)&Bs[1][rA][cA] };
    // Running per-lane global pointers (advance 32 bf16 = 64B per K-tile).
    const __bf16* pA0 = A + (size_t)(m0 + rA) * lda + cA;
    const __bf16* pA1 = A + (size_t)(m0 + rA + 64) * lda + cA;
    const __bf16* pB  = W + (size_t)(n0 + rA) * K + cA;

    // Prologue: DMA tile 0 into buffer 0.
    ASYNC_B128(ldsA0[0], (unsigned long long)pA0);
    ASYNC_B128(ldsA1[0], (unsigned long long)pA1);
    ASYNC_B128(ldsB[0],  (unsigned long long)pB);

    v8f acc[2][2];
#pragma unroll
    for (int i = 0; i < 2; ++i)
#pragma unroll
        for (int j = 0; j < 2; ++j)
#pragma unroll
            for (int v = 0; v < 8; ++v) acc[i][j][v] = 0.f;

    const int nIter = K >> 5;
    // Steady state: branch-free issue of tile it+1, compute tile it.
    for (int it = 0; it < nIter - 1; ++it) {
        const int cur = it & 1;
        const int nxt = cur ^ 1;
        pA0 += 32; pA1 += 32; pB += 32;
        ASYNC_B128(ldsA0[nxt], (unsigned long long)pA0);
        ASYNC_B128(ldsA1[nxt], (unsigned long long)pA1);
        ASYNC_B128(ldsB[nxt],  (unsigned long long)pB);
        // <=3 outstanding: current tile's 3 DMAs retired (in-order completion).
        asm volatile("s_wait_asynccnt 0x3" ::: "memory");
        __syncthreads();
        gemm_tile_compute(&As[cur][0][0], &Bs[cur][0][0], wm, wn, lane, acc);
        __syncthreads();   // protects the buffer the next iteration DMAs into
    }
    // Epilogue: last tile.
    asm volatile("s_wait_asynccnt 0x0" ::: "memory");
    __syncthreads();
    gemm_tile_compute(&As[(nIter - 1) & 1][0][0], &Bs[(nIter - 1) & 1][0][0],
                      wm, wn, lane, acc);

    const int rofs = (lane >> 4) ? 8 : 0;
    const int cn   = lane & 15;
#pragma unroll
    for (int mi = 0; mi < 2; ++mi)
#pragma unroll
        for (int ni = 0; ni < 2; ++ni) {
            int col = n0 + wn * 32 + ni * 16 + cn;
#pragma unroll
            for (int v = 0; v < 8; ++v) {
                int row = m0 + wm * 32 + mi * 16 + v + rofs;
                C[(size_t)row * N + col] = acc[mi][ni][v];
            }
        }
}

// ---------------------------------------------------------------------------
// Row-wise RMS norm (in place). One 256-thread block per row.
// ---------------------------------------------------------------------------
__global__ __launch_bounds__(256) void rmsnorm_kernel(
    float* __restrict__ x, const float* __restrict__ w, int R, int stride) {
    __shared__ float red[8];
    float* p = x + (size_t)blockIdx.x * stride;
    float ss = 0.f;
    for (int i = threadIdx.x; i < R; i += 256) { float v = p[i]; ss += v * v; }
#pragma unroll
    for (int o = 16; o; o >>= 1) ss += __shfl_xor(ss, o, 32);
    if ((threadIdx.x & 31) == 0) red[threadIdx.x >> 5] = ss;
    __syncthreads();
    float tot = 0.f;
#pragma unroll
    for (int k = 0; k < 8; ++k) tot += red[k];
    float scale = rsqrtf(tot / (float)R + 1e-6f);
    for (int i = threadIdx.x; i < R; i += 256) p[i] = p[i] * scale * w[i];
}

// ---------------------------------------------------------------------------
// RoPE for q (in place): q is (tokens, H*192); rotate dims [h*192+128, +191].
// ---------------------------------------------------------------------------
__global__ __launch_bounds__(512) void rope_q_kernel(
    float* __restrict__ q, const float* __restrict__ fr) {
    int token = blockIdx.x;
    int pr = threadIdx.x;
    int h = pr >> 5, i = pr & 31;
    int s = token & (MLA_S - 1);
    float c  = fr[(s * 32 + i) * 2];
    float sn = fr[(s * 32 + i) * 2 + 1];
    float* p = q + (size_t)token * (MLA_H * 192) + h * 192 + 128 + 2 * i;
    float x0 = p[0], x1 = p[1];
    p[0] = x0 * c - x1 * sn;
    p[1] = x0 * sn + x1 * c;
}

// ---------------------------------------------------------------------------
// RoPE for k_pe: read kv[token, 512+2i..+1] (stride 576) -> kpe[token, 64].
// ---------------------------------------------------------------------------
__global__ __launch_bounds__(128) void rope_k_kernel(
    const float* __restrict__ kv, const float* __restrict__ fr,
    float* __restrict__ kpe) {
    int token = blockIdx.x * 4 + (threadIdx.x >> 5);
    int i = threadIdx.x & 31;
    int s = token & (MLA_S - 1);
    float c  = fr[(s * 32 + i) * 2];
    float sn = fr[(s * 32 + i) * 2 + 1];
    const float* p = kv + (size_t)token * 576 + 512 + 2 * i;
    float x0 = p[0], x1 = p[1];
    kpe[(size_t)token * 64 + 2 * i]     = x0 * c - x1 * sn;
    kpe[(size_t)token * 64 + 2 * i + 1] = x0 * sn + x1 * c;
}

// ---------------------------------------------------------------------------
// Causal flash attention per (b, h, q-tile of 64). Block 256 threads = 8 waves.
// Q: (tokens, H*192) RoPE'd; KVB: (tokens, H*256) k_nope|v; KPE: (tokens, 64);
// O: (tokens, H*128). K-tiles of 32; scores 4x2 wave grid; P*V 4 accs/wave.
// ---------------------------------------------------------------------------
__global__ __launch_bounds__(256) void mla_attn(
    const float* __restrict__ Q, const float* __restrict__ KVB,
    const float* __restrict__ KPE, float* __restrict__ O) {
    __shared__ __bf16 Qs[64][192];
    __shared__ __bf16 Ks[32][192];
    __shared__ __bf16 Vt[128][32];   // transposed: [d][key]
    __shared__ float  Ss[64][32];
    __shared__ __bf16 Ps[64][32];
    __shared__ float  mrow[64], lrow[64], arow[64];

    const int tid  = threadIdx.x;
    const int lane = tid & 31;
    const int wave = tid >> 5;
    const int wm   = wave >> 1;
    const int wn   = wave & 1;
    const int qt   = blockIdx.x;
    const int b    = blockIdx.y >> 4;
    const int h    = blockIdx.y & 15;
    const int q0   = qt * 64;
    const size_t tok0 = (size_t)b * MLA_S;

#pragma unroll
    for (int i = 0; i < 12; ++i) {
        int f = tid + i * 256;
        int r = f / 48, c = (f % 48) * 4;
        float4 v = *(const float4*)(Q + ((tok0 + q0 + r) * MLA_H + h) * 192 + c);
        *(bf16x4*)&Qs[r][c] = cvt4(v);
    }
    if (tid < 64) { mrow[tid] = -3e38f; lrow[tid] = 0.f; }

    v8f oacc[4];
#pragma unroll
    for (int j = 0; j < 4; ++j)
#pragma unroll
        for (int v = 0; v < 8; ++v) oacc[j][v] = 0.f;

    const int nkt = 2 * qt + 2;
    for (int kt2 = 0; kt2 < nkt; ++kt2) {
        const int kb = kt2 * 32;
#pragma unroll
        for (int i = 0; i < 4; ++i) {
            int f = tid + i * 256;
            int r = f >> 5, c = (f & 31) * 4;
            float4 v = *(const float4*)(KVB + ((tok0 + kb + r) * MLA_H + h) * 256 + c);
            *(bf16x4*)&Ks[r][c] = cvt4(v);
        }
#pragma unroll
        for (int i = 0; i < 2; ++i) {
            int f = tid + i * 256;
            int r = f >> 4, c = (f & 15) * 4;
            float4 v = *(const float4*)(KPE + (tok0 + kb + r) * 64 + c);
            *(bf16x4*)&Ks[r][128 + c] = cvt4(v);
        }
#pragma unroll
        for (int i = 0; i < 4; ++i) {
            int f = tid + i * 256;
            int r = f >> 5, c = (f & 31) * 4;
            float4 v = *(const float4*)(KVB + ((tok0 + kb + r) * MLA_H + h) * 256 + 128 + c);
            Vt[c + 0][r] = (__bf16)v.x;
            Vt[c + 1][r] = (__bf16)v.y;
            Vt[c + 2][r] = (__bf16)v.z;
            Vt[c + 3][r] = (__bf16)v.w;
        }
        __syncthreads();

        v8f s;
#pragma unroll
        for (int v = 0; v < 8; ++v) s[v] = 0.f;
#pragma unroll
        for (int ks = 0; ks < 192; ks += 32) {
            v16bf a  = load_a_frag(&Qs[0][0], 192, wm * 16, ks, lane);
            v16bf bf = load_b_frag(&Ks[0][0], 192, wn * 16, ks, lane);
            s = wmma_bf16(a, bf, s);
        }
        {
            const int rofs = (lane >> 4) ? 8 : 0;
            const int cn   = lane & 15;
#pragma unroll
            for (int v = 0; v < 8; ++v)
                Ss[wm * 16 + v + rofs][wn * 16 + cn] = s[v];
        }
        __syncthreads();

        {
            int r  = tid >> 2, qq = tid & 3;
            int qi = q0 + r;
            float mx = -3e38f;
#pragma unroll
            for (int jj = 0; jj < 8; ++jj) {
                int j = qq * 8 + jj;
                if (kb + j <= qi) mx = fmaxf(mx, Ss[r][j] * MLA_SCALE);
            }
            mx = fmaxf(mx, __shfl_xor(mx, 1, 32));
            mx = fmaxf(mx, __shfl_xor(mx, 2, 32));
            float mold = mrow[r];
            float mnew = fmaxf(mold, mx);
            float aa   = __expf(mold - mnew);
            float sum  = 0.f;
#pragma unroll
            for (int jj = 0; jj < 8; ++jj) {
                int j = qq * 8 + jj;
                float pv = 0.f;
                if (kb + j <= qi) pv = __expf(Ss[r][j] * MLA_SCALE - mnew);
                Ps[r][j] = (__bf16)pv;
                sum += pv;
            }
            sum += __shfl_xor(sum, 1, 32);
            sum += __shfl_xor(sum, 2, 32);
            if (qq == 0) {
                mrow[r] = mnew;
                arow[r] = aa;
                lrow[r] = lrow[r] * aa + sum;
            }
        }
        __syncthreads();

        {
            const int rofs = (lane >> 4) ? 8 : 0;
            float al[8];
#pragma unroll
            for (int v = 0; v < 8; ++v) al[v] = arow[wm * 16 + v + rofs];
#pragma unroll
            for (int j = 0; j < 4; ++j)
#pragma unroll
                for (int v = 0; v < 8; ++v) oacc[j][v] *= al[v];
            v16bf a = load_a_frag(&Ps[0][0], 32, wm * 16, 0, lane);
#pragma unroll
            for (int j = 0; j < 4; ++j) {
                v16bf bf = load_b_frag(&Vt[0][0], 32, (wn * 4 + j) * 16, 0, lane);
                oacc[j] = wmma_bf16(a, bf, oacc[j]);
            }
        }
        __syncthreads();
    }

    const int rofs = (lane >> 4) ? 8 : 0;
    const int cn   = lane & 15;
#pragma unroll
    for (int j = 0; j < 4; ++j) {
        int col = h * 128 + (wn * 4 + j) * 16 + cn;
#pragma unroll
        for (int v = 0; v < 8; ++v) {
            int row = wm * 16 + v + rofs;
            O[(tok0 + q0 + row) * (MLA_H * 128) + col] = oacc[j][v] / lrow[row];
        }
    }
}

// ---------------------------------------------------------------------------
extern "C" void kernel_launch(void* const* d_in, const int* in_sizes, int n_in,
                              void* d_out, int out_size, void* d_ws, size_t ws_size,
                              hipStream_t stream) {
    const float* x     = (const float*)d_in[0];   // (2,2048,2048)
    const float* freqs = (const float*)d_in[1];   // (2048,32,2)
    const float* Wqa   = (const float*)d_in[2];   // (1536,2048)
    const float* qln   = (const float*)d_in[3];   // (1536,)
    const float* Wqb   = (const float*)d_in[4];   // (3072,1536)
    const float* Wkva  = (const float*)d_in[5];   // (576,2048)
    const float* kvln  = (const float*)d_in[6];   // (512,)
    const float* Wkvb  = (const float*)d_in[7];   // (4096,512)
    const float* Wo    = (const float*)d_in[8];   // (2048,2048)
    float* out = (float*)d_out;                   // (4096,2048)

    const int M = 2 * MLA_S;                      // 4096 tokens

    // ---- f32 scratch ----
    float* wsf  = (float*)d_ws;
    float* qa   = wsf;                            // M*1536
    float* qb   = qa  + (size_t)M * 1536;         // M*3072
    float* kv   = qb  + (size_t)M * 3072;         // M*576
    float* kpe  = kv  + (size_t)M * 576;          // M*64
    float* kvb  = kpe + (size_t)M * 64;           // M*4096
    float* aout = kvb + (size_t)M * 4096;         // M*2048
    // ---- bf16 scratch (after all f32 buffers) ----
    __bf16* wsb   = (__bf16*)(aout + (size_t)M * 2048);
    __bf16* xb    = wsb;                          // M*2048
    __bf16* qab   = xb    + (size_t)M * 2048;     // M*1536
    __bf16* kvcb  = qab   + (size_t)M * 1536;     // M*512
    __bf16* aob   = kvcb  + (size_t)M * 512;      // M*2048
    __bf16* Wqab  = aob   + (size_t)M * 2048;     // 1536*2048
    __bf16* Wqbb  = Wqab  + (size_t)1536 * 2048;  // 3072*1536
    __bf16* Wkvab = Wqbb  + (size_t)3072 * 1536;  // 576*2048
    __bf16* Wkvbb = Wkvab + (size_t)576 * 2048;   // 4096*512
    __bf16* Wob   = Wkvbb + (size_t)4096 * 512;   // 2048*2048

    dim3 blk(256);
    // Pre-convert operands to bf16 (halves HBM traffic; enables async DMA).
    cvt_f32_bf16<<<(size_t)M * 2048 / 1024, blk, 0, stream>>>(x, xb);
    cvt_f32_bf16<<<(size_t)1536 * 2048 / 1024, blk, 0, stream>>>(Wqa, Wqab);
    cvt_f32_bf16<<<(size_t)3072 * 1536 / 1024, blk, 0, stream>>>(Wqb, Wqbb);
    cvt_f32_bf16<<<(size_t)576 * 2048 / 1024, blk, 0, stream>>>(Wkva, Wkvab);
    cvt_f32_bf16<<<(size_t)4096 * 512 / 1024, blk, 0, stream>>>(Wkvb, Wkvbb);
    cvt_f32_bf16<<<(size_t)2048 * 2048 / 1024, blk, 0, stream>>>(Wo, Wob);

    // q_a = x @ Wqa^T ; kv = x @ Wkva^T
    gemm_bf16_async<<<dim3(1536 / 64, M / 128), blk, 0, stream>>>(xb, 2048, Wqab, qa, M, 1536, 2048);
    gemm_bf16_async<<<dim3(576 / 64,  M / 128), blk, 0, stream>>>(xb, 2048, Wkvab, kv, M, 576, 2048);
    // RMS norms (in place, f32)
    rmsnorm_kernel<<<M, 256, 0, stream>>>(qa, qln, 1536, 1536);
    rmsnorm_kernel<<<M, 256, 0, stream>>>(kv, kvln, 512, 576);
    // k_pe RoPE -> kpe (f32)
    rope_k_kernel<<<M / 4, 128, 0, stream>>>(kv, freqs, kpe);
    // q = q_a_norm @ Wqb^T, then per-head RoPE (in place, f32)
    cvt_f32_bf16<<<(size_t)M * 1536 / 1024, blk, 0, stream>>>(qa, qab);
    gemm_bf16_async<<<dim3(3072 / 64, M / 128), blk, 0, stream>>>(qab, 1536, Wqbb, qb, M, 3072, 1536);
    rope_q_kernel<<<M, 512, 0, stream>>>(qb, freqs);
    // kvb = kv_c_norm @ Wkvb^T  (compact 576-stride rows to 512 while converting)
    cvt_rows_bf16<<<M, 128, 0, stream>>>(kv, 576, kvcb, 512);
    gemm_bf16_async<<<dim3(4096 / 64, M / 128), blk, 0, stream>>>(kvcb, 512, Wkvbb, kvb, M, 4096, 512);
    // causal flash attention -> aout (tokens, H*128)
    mla_attn<<<dim3(MLA_S / 64, 2 * MLA_H), blk, 0, stream>>>(qb, kvb, kpe, aout);
    // out = aout @ Wo^T
    cvt_f32_bf16<<<(size_t)M * 2048 / 1024, blk, 0, stream>>>(aout, aob);
    gemm_bf16_async<<<dim3(2048 / 64, M / 128), blk, 0, stream>>>(aob, 2048, Wob, out, M, 2048, 2048);
}
